// EfficientKANConv2D_68032281969068
// MI455X (gfx1250) — matrix operand order repaired and checked by reading, not compiled, not run on hardware
//
#include <hip/hip_runtime.h>
#include <hip/hip_bf16.h>
#include <math.h>

typedef __attribute__((ext_vector_type(16))) _Float16 v16h;
typedef __attribute__((ext_vector_type(8)))  _Float16 v8h;
typedef __attribute__((ext_vector_type(8)))  float    v8f;
typedef __attribute__((ext_vector_type(4)))  unsigned int uint4v;

#define GRIDSZ 5
#define ORD 3
#define NG 12            // grid points = GRIDSZ + 2*ORD + 1
#define NS 8             // GRIDSZ + ORD basis functions
#define CIN 16
#define COUT 32
#define BB 4
#define HH 128
#define WW 128
#define HW (HH * WW)
#define HP 130           // H + 2 halo
#define WP 130
#define KEPS 1e-8f

// ---------------- workspace layout (in _Float16 elements) ----------------
// basisWs: [B][HP][WP][128]  f16   (c' = c*8 + s, channel-innermost)
// xWs:     [B][HP][WP][32]   f16   (ch 0..15 = x, ch 16..31 = zero pad)
// wS:      [mt2][tap9][kk4][lane32][e16] f16  A-fragments, spline
// wB:      [mt2][tap9][lane32][e16]      f16  A-fragments, base (K padded to 32)
#define BASIS_HALVES (BB * HP * WP * 128)     // 8,652,800
#define XWS_HALVES   (BB * HP * WP * 32)      // 2,163,200
#define WS_HALVES    (2 * 9 * 4 * 32 * 16)    // 36,864
#define WB_HALVES    (2 * 9 * 32 * 16)        // 9,216
#define ZERO_BYTES   ((BASIS_HALVES + XWS_HALVES) * 2)   // 21,632,000 (mult of 16)

// ---------------- kernel 1: zero activation workspaces -------------------
__global__ __launch_bounds__(256) void kan_fill(uint4v* __restrict__ p, int n16) {
    uint4v z = {0u, 0u, 0u, 0u};
    for (int i = blockIdx.x * 256 + threadIdx.x; i < n16; i += gridDim.x * 256)
        p[i] = z;
}

// ---------------- kernel 2: B-spline basis expansion ----------------------
__global__ __launch_bounds__(256) void kan_expand(
    const float* __restrict__ x, const float* __restrict__ grid,
    _Float16* __restrict__ basisWs, _Float16* __restrict__ xWs)
{
    int tid = blockIdx.x * 256 + threadIdx.x;   // [0, B*H*W*CIN)
    int c = tid & 15;
    int w = (tid >> 4) & 127;
    int h = (tid >> 11) & 127;
    int b = tid >> 18;

    float xv = x[((b * CIN + c) * HH + h) * WW + w];

    float gr[NG];
#pragma unroll
    for (int i = 0; i < NG; ++i) gr[i] = grid[i];

    float bas[NG - 1];
#pragma unroll
    for (int j = 0; j < NG - 1; ++j)
        bas[j] = (xv >= gr[j] && xv < gr[j + 1]) ? 1.0f : 0.0f;
#pragma unroll
    for (int k = 1; k <= ORD; ++k) {
#pragma unroll
        for (int j = 0; j < NG - 1 - k; ++j) {
            float left  = (xv - gr[j]) / (gr[j + k] - gr[j] + KEPS);
            float right = (gr[j + k + 1] - xv) / (gr[j + k + 1] - gr[j + 1] + KEPS);
            bas[j] = left * bas[j] + right * bas[j + 1];
        }
    }

    int P = (b * HP + (h + 1)) * WP + (w + 1);  // halo'd pixel index
    v8h o8;
#pragma unroll
    for (int s = 0; s < NS; ++s) o8[s] = (_Float16)bas[s];
    *(v8h*)(basisWs + (size_t)P * 128 + c * 8) = o8;
    xWs[(size_t)P * 32 + c] = (_Float16)xv;
}

// ---------------- kernel 3: swizzle weights into A-fragment layout --------
// lane L: M = L%16 (+16*mt); half-slot e -> v=e/2,d=e%2;
// kwithin = 16*(L/16) + e                              (contiguous k order, matches B)
__global__ __launch_bounds__(256) void kan_prep(
    const float* __restrict__ baseW, const float* __restrict__ splineW,
    _Float16* __restrict__ wS, _Float16* __restrict__ wB)
{
    int tid = blockIdx.x * 256 + threadIdx.x;
    if (tid < WS_HALVES) {
        int e   = tid & 15;
        int L   = (tid >> 4) & 31;
        int kk  = (tid >> 9) & 3;
        int tap = (tid >> 11) % 9;
        int mt  = tid / (2048 * 9);
        int m = (L & 15) + 16 * mt;
        int g = L >> 4;
        int v = e >> 1, d = e & 1;
        int kwithin = 16 * g + e;
        int cp = kk * 32 + kwithin;           // c' in [0,128)
        int c = cp >> 3, s = cp & 7;
        int kh = tap / 3, kw = tap % 3;
        float val = splineW[((((m * CIN + c) * 3 + kh) * 3 + kw) * NS) + s];
        wS[tid] = (_Float16)val;
    } else if (tid < WS_HALVES + WB_HALVES) {
        int t = tid - WS_HALVES;
        int e   = t & 15;
        int L   = (t >> 4) & 31;
        int tap = (t >> 9) % 9;
        int mt  = t / (512 * 9);
        int m = (L & 15) + 16 * mt;
        int g = L >> 4;
        int v = e >> 1, d = e & 1;
        int kwithin = 16 * g + e;
        int kh = tap / 3, kw = tap % 3;
        float val = (kwithin < CIN)
                  ? baseW[((m * CIN + kwithin) * 3 + kh) * 3 + kw] : 0.0f;
        wB[t] = (_Float16)val;
    }
}

// ---------------- kernel 4: implicit-GEMM conv via WMMA -------------------
// wave tile: M=32 (2 wmma M-tiles) x N=16 pixels (one row segment).
// 8 waves per block -> 8 tiles; grid = 4096/8 = 512 blocks.
// All in-loop addresses = per-lane base pointer + compile-time immediate.
__global__ __launch_bounds__(256) void kan_gemm(
    const _Float16* __restrict__ basisWs, const _Float16* __restrict__ xWs,
    const _Float16* __restrict__ wS, const _Float16* __restrict__ wB,
    float* __restrict__ out)
{
    const int lane = threadIdx.x & 31;
    const int wave = threadIdx.x >> 5;
    const int tile = blockIdx.x * 8 + wave;   // 0..4095
    const int wt = tile & 7;
    const int h  = (tile >> 3) & 127;
    const int b  = tile >> 10;
    const int w0 = wt * 16;
    const int n = lane & 15, g = lane >> 4;

    // per-lane base pointers (tap (0,0) halo'd pixel); all tap/kk deltas are constants
    const int P0 = (b * HP + h) * WP + (w0 + n);
    const _Float16* __restrict__ pbasis = basisWs + (size_t)P0 * 128 + g * 16;
    const _Float16* __restrict__ px     = xWs     + (size_t)P0 * 32  + g * 16;
    const _Float16* __restrict__ pwS    = wS + lane * 16;
    const _Float16* __restrict__ pwB    = wB + lane * 16;

    v8f aS0 = {}, aS1 = {}, aB0 = {}, aB1 = {};

#pragma unroll
    for (int kh = 0; kh < 3; ++kh) {
#pragma unroll
        for (int kw = 0; kw < 3; ++kw) {
            const int tap = kh * 3 + kw;
            const int dpx = kh * WP + kw;     // pixel delta (compile-time const)

            // ---- base branch: K=32 step (upper 16 channels are zero pad)
            v16h bb = *(const v16h*)(px + dpx * 32);
            v16h a0 = *(const v16h*)(pwB + (0 * 9 + tap) * 512);
            v16h a1 = *(const v16h*)(pwB + (1 * 9 + tap) * 512);
            aB0 = __builtin_amdgcn_wmma_f32_16x16x32_f16(false, a0, false, bb,
                                                         (short)0, aB0, false, false);
            aB1 = __builtin_amdgcn_wmma_f32_16x16x32_f16(false, a1, false, bb,
                                                         (short)0, aB1, false, false);

            // ---- spline branch: 4 K-steps of 32 over 128 basis channels
#pragma unroll
            for (int kk = 0; kk < 4; ++kk) {
                v16h bs = *(const v16h*)(pbasis + dpx * 128 + kk * 32);
                v16h s0 = *(const v16h*)(pwS + (0 * 36 + tap * 4 + kk) * 512);
                v16h s1 = *(const v16h*)(pwS + (1 * 36 + tap * 4 + kk) * 512);
                aS0 = __builtin_amdgcn_wmma_f32_16x16x32_f16(false, s0, false, bs,
                                                             (short)0, aS0, false, false);
                aS1 = __builtin_amdgcn_wmma_f32_16x16x32_f16(false, s1, false, bs,
                                                             (short)0, aS1, false, false);
            }
        }
    }

    // epilogue: out = silu(base) + spline.
    // D layout: col = lane%16, VGPR r -> row M = r + 8*(lane/16); mt=1 adds 16.
    float* __restrict__ outp =
        out + (((size_t)b * COUT + 8 * g) * HH + h) * WW + (w0 + n);
#pragma unroll
    for (int r = 0; r < 8; ++r) {
        float zb0 = aB0[r];
        outp[r * HW] = zb0 / (1.0f + __expf(-zb0)) + aS0[r];
        float zb1 = aB1[r];
        outp[(r + 16) * HW] = zb1 / (1.0f + __expf(-zb1)) + aS1[r];
    }
}

// ---------------- host launcher ------------------------------------------
extern "C" void kernel_launch(void* const* d_in, const int* in_sizes, int n_in,
                              void* d_out, int out_size, void* d_ws, size_t ws_size,
                              hipStream_t stream) {
    const float* x    = (const float*)d_in[0];
    const float* bw   = (const float*)d_in[1];
    const float* sw   = (const float*)d_in[2];
    const float* grid = (const float*)d_in[3];
    float* out = (float*)d_out;

    _Float16* ws      = (_Float16*)d_ws;
    _Float16* basisWs = ws;
    _Float16* xWs     = basisWs + BASIS_HALVES;
    _Float16* wS      = xWs + XWS_HALVES;
    _Float16* wB      = wS + WS_HALVES;

    // 1) zero halo'd activation workspaces (16B vector fill)
    kan_fill<<<2048, 256, 0, stream>>>((uint4v*)d_ws, ZERO_BYTES / 16);

    // 2) swizzle weights into WMMA A-fragment layout (46080 threads)
    kan_prep<<<(WS_HALVES + WB_HALVES + 255) / 256, 256, 0, stream>>>(bw, sw, wS, wB);

    // 3) basis expansion: B*H*W*CIN = 1,048,576 threads
    kan_expand<<<(BB * HH * WW * CIN) / 256, 256, 0, stream>>>(x, grid, basisWs, xWs);

    // 4) WMMA implicit-GEMM conv: 4096 wave-tiles / 8 waves per block
    kan_gemm<<<512, 256, 0, stream>>>(basisWs, xWs, wS, wB, out);
}
